// Word_GNN_30528627539985
// MI455X (gfx1250) — matrix operand order repaired and checked by reading, not compiled, not run on hardware
//
#include <hip/hip_runtime.h>

typedef __attribute__((ext_vector_type(16))) __bf16 v16bf;
typedef __attribute__((ext_vector_type(8)))  float  v8f;
typedef unsigned short u16;

#define NNODE 96
#define DDIM  768
#define HDIM  64
#define BATCH 4

__device__ __forceinline__ u16 f2bf(float x) {
  unsigned u = __float_as_uint(x);
  u += 0x7FFFu + ((u >> 16) & 1u);           // round-to-nearest-even
  return (u16)(u >> 16);
}
__device__ __forceinline__ float bf2f(u16 h) {
  return __uint_as_float(((unsigned)h) << 16);
}

// ---- WMMA fragment helpers (ISA 7.12.2 layouts, 16x16x32 bf16) -------------
// A (MxK row-major): lane holds row m = mrow + (lane&15);
//   halves 0..7  at k = kb + 8*(lane>>4), halves 8..15 at +16
__device__ __forceinline__ v16bf load_a_frag(const u16* A, int ld, int mrow,
                                             int kb, int lane) {
  const u16* p = A + (size_t)(mrow + (lane & 15)) * ld + kb + ((lane >> 4) << 3);
  union { v16bf v; uint4 q[2]; } u;
  u.q[0] = *(const uint4*)(p);
  u.q[1] = *(const uint4*)(p + 16);
  return u.v;
}
// B source stored as (N x K) row-major (weight W[out,in]); computes X@W^T.
// lane holds column n = ncol + (lane&15); halves 0..7 at
// k = kb + 8*(lane>>4), halves 8..15 at +16.
__device__ __forceinline__ v16bf load_b_frag(const u16* B, int ld, int ncol,
                                             int kb, int lane) {
  const u16* p = B + (size_t)(ncol + (lane & 15)) * ld + kb + ((lane >> 4) << 3);
  union { v16bf v; uint4 q[2]; } u;
  u.q[0] = *(const uint4*)(p);
  u.q[1] = *(const uint4*)(p + 16);
  return u.v;
}
__device__ __forceinline__ v8f wmma_bf16(v16bf a, v16bf b, v8f c) {
  return __builtin_amdgcn_wmma_f32_16x16x32_bf16(false, a, false, b,
                                                 (short)0, c, false, false);
}

// ---- generic GEMM: C(MxN) = A(MxK)bf16 @ B(NxK)bf16^T + bias, act ----------
// 2x2 register blocking: one wave owns a 32x32 macro-tile -> every A/B
// fragment feeds two WMMAs (4 loads : 4 wmma per K-step).
__global__ void __launch_bounds__(128)
k_gemm(const u16* __restrict__ A, const u16* __restrict__ B,
       const float* __restrict__ bias, float* __restrict__ Cf,
       u16* __restrict__ Ch, int M, int N, int K, int act) {
  int lane = threadIdx.x & 31;
  int tile = blockIdx.x * (blockDim.x >> 5) + (threadIdx.x >> 5);
  int nmac = N >> 5;
  int tot = (M >> 5) * nmac;
  if (tile >= tot) return;                   // uniform per wave
  int tm = (tile / nmac) * 32, tn = (tile % nmac) * 32;
  v8f acc00 = {}, acc01 = {}, acc10 = {}, acc11 = {};
  for (int kb = 0; kb < K; kb += 32) {
    if (kb + 64 < K) {                       // global_prefetch_b8, 2 steps ahead
      __builtin_prefetch(A + (size_t)(tm + (lane & 15)) * K + kb + 64, 0, 1);
      __builtin_prefetch(B + (size_t)(tn + (lane & 15)) * K + kb + 64, 0, 1);
    }
    v16bf a0 = load_a_frag(A, K, tm,      kb, lane);
    v16bf a1 = load_a_frag(A, K, tm + 16, kb, lane);
    v16bf b0 = load_b_frag(B, K, tn,      kb, lane);
    v16bf b1 = load_b_frag(B, K, tn + 16, kb, lane);
    acc00 = wmma_bf16(a0, b0, acc00);
    acc01 = wmma_bf16(a0, b1, acc01);
    acc10 = wmma_bf16(a1, b0, acc10);
    acc11 = wmma_bf16(a1, b1, acc11);
  }
  v8f* accs[2][2] = {{&acc00, &acc01}, {&acc10, &acc11}};
#pragma unroll
  for (int mi = 0; mi < 2; ++mi) {
#pragma unroll
    for (int ni = 0; ni < 2; ++ni) {
      int n  = tn + ni * 16 + (lane & 15);
      int mb = tm + mi * 16 + ((lane >> 4) << 3);
      float bv = bias ? bias[n] : 0.f;
      v8f a = *accs[mi][ni];
#pragma unroll
      for (int v = 0; v < 8; ++v) {
        float x = a[v] + bv;
        if (act) x = fmaxf(x, 0.f);
        size_t off = (size_t)(mb + v) * N + n;
        if (Cf) Cf[off] = x;
        if (Ch) Ch[off] = f2bf(x);
      }
    }
  }
}

// ---- cosine-similarity gram: sem_adj[b] = S_b S_b^T (S normalized rows) ----
__global__ void __launch_bounds__(128)
k_gram(const u16* __restrict__ sem, float* __restrict__ sem_adj) {
  int lane = threadIdx.x & 31;
  int tile = blockIdx.x * (blockDim.x >> 5) + (threadIdx.x >> 5); // 0..143
  int b = tile / 36, t = tile % 36;
  int tm = t / 6, tn = t % 6;
  const u16* S = sem + (size_t)b * NNODE * DDIM;
  v8f acc = {};
  for (int kb = 0; kb < DDIM; kb += 32) {
    v16bf a = load_a_frag(S, DDIM, tm * 16, kb, lane);
    v16bf bb = load_b_frag(S, DDIM, tn * 16, kb, lane);
    acc = wmma_bf16(a, bb, acc);
  }
  float* C = sem_adj + (size_t)b * NNODE * NNODE;
  int n  = tn * 16 + (lane & 15);
  int mb = tm * 16 + ((lane >> 4) << 3);
#pragma unroll
  for (int v = 0; v < 8; ++v) C[(size_t)(mb + v) * NNODE + n] = acc[v];
}

// ---- row-normalize feature -> bf16 sem ------------------------------------
__global__ void __launch_bounds__(256)
k_sem_rows(const float* __restrict__ F, u16* __restrict__ sem) {
  __shared__ float red[256];
  int row = blockIdx.x, tid = threadIdx.x;
  const float* x = F + (size_t)row * DDIM;
  float s = 0.f;
  for (int d = tid; d < DDIM; d += 256) { float v = x[d]; s += v * v; }
  red[tid] = s; __syncthreads();
  for (int o = 128; o > 0; o >>= 1) {
    if (tid < o) red[tid] += red[tid + o];
    __syncthreads();
  }
  float inv = 1.f / fmaxf(sqrtf(red[0]), 1e-12f);
  for (int d = tid; d < DDIM; d += 256)
    sem[(size_t)row * DDIM + d] = f2bf(x[d] * inv);
}

// ---- EK = E_dep @ W_k^T  (50 x 768), fp32 ----------------------------------
__global__ void __launch_bounds__(256)
k_ek(const float* __restrict__ E, const float* __restrict__ Wk,
     float* __restrict__ EK) {
  int t = blockIdx.x * 256 + threadIdx.x;
  if (t >= 50 * DDIM) return;
  int r = t / DDIM, d = t % DDIM;
  const float* e = E + (size_t)r * DDIM;
  const float* w = Wk + (size_t)d * DDIM;
  float s = 0.f;
  for (int k = 0; k < DDIM; ++k) s += e[k] * w[k];
  EK[t] = s;
}

__global__ void k_cvt_bf16(const float* __restrict__ s, u16* __restrict__ d,
                           int n) {
  int t = blockIdx.x * 256 + threadIdx.x;
  if (t < n) d[t] = f2bf(s[t]);
}

// ---- heavy kernel: one block per (b,i); gated msg GEMM -> LN -> softmax -> rf
__global__ void __launch_bounds__(256)
k_attn_rf(const u16* __restrict__ tt_h, const float* __restrict__ Epos,
          const int* __restrict__ oneshot, const int* __restrict__ rel_adj,
          const float* __restrict__ Edep, const u16* __restrict__ Wa_h,
          const float* __restrict__ b_a, const float* __restrict__ ln_g,
          const float* __restrict__ ln_b, const float* __restrict__ W_d,
          const float* __restrict__ b_d, const float* __restrict__ rel_mask,
          const float* __restrict__ sem_adj, const float* __restrict__ F32,
          float* __restrict__ rf32, u16* __restrict__ rf_h) {
  const int N = NNODE, D = DDIM, H = HDIM;
  int b = blockIdx.x / N, i = blockIdx.x % N;
  int tid = threadIdx.x, lane = tid & 31, wave = tid >> 5;
  __shared__ __align__(16) u16 Ash[NNODE * 32];     // gated A tile (96x32)
  __shared__ float Csh[NNODE * HDIM];               // a-matrix (96x64)
  __shared__ float logits[NNODE];
  __shared__ float sred[2];
  const float* pos = Epos + (size_t)oneshot[b * N + i] * D;
  const int* ra = rel_adj + ((size_t)b * N + i) * N;
  const u16* ttb = tt_h + (size_t)b * N * D;
  v8f acc0 = {}, acc1 = {}, acc2 = {};
  for (int kb = 0; kb < D; kb += 32) {
    for (int e = tid; e < N * 32; e += 256) {
      int j = e >> 5, k = (e & 31) + kb;
      float ev = Edep[(size_t)ra[j] * D + k];
      float g  = 1.f / (1.f + __expf(-(bf2f(ttb[(size_t)j * D + k]) + pos[k])));
      Ash[e] = f2bf(ev * g);
    }
    __syncthreads();
    int job = wave;                                  // 24 jobs = 6 jt x 4 ht
    v16bf a = load_a_frag(Ash, 32, (job >> 2) * 16, 0, lane);
    v16bf bf = load_b_frag(Wa_h, D, (job & 3) * 16, kb, lane);
    acc0 = wmma_bf16(a, bf, acc0);
    job = wave + 8;
    a  = load_a_frag(Ash, 32, (job >> 2) * 16, 0, lane);
    bf = load_b_frag(Wa_h, D, (job & 3) * 16, kb, lane);
    acc1 = wmma_bf16(a, bf, acc1);
    job = wave + 16;
    a  = load_a_frag(Ash, 32, (job >> 2) * 16, 0, lane);
    bf = load_b_frag(Wa_h, D, (job & 3) * 16, kb, lane);
    acc2 = wmma_bf16(a, bf, acc2);
    __syncthreads();
  }
  {
    int n = lane & 15, mb = (lane >> 4) << 3;
    int j0 = (wave >> 2) * 16, h0 = (wave & 3) * 16;
#pragma unroll
    for (int v = 0; v < 8; ++v) Csh[(j0 + mb + v) * H + h0 + n] = acc0[v];
    int jb = ((wave + 8) >> 2) * 16, hb = ((wave + 8) & 3) * 16;
#pragma unroll
    for (int v = 0; v < 8; ++v) Csh[(jb + mb + v) * H + hb + n] = acc1[v];
    int jc = ((wave + 16) >> 2) * 16, hc = ((wave + 16) & 3) * 16;
#pragma unroll
    for (int v = 0; v < 8; ++v) Csh[(jc + mb + v) * H + hc + n] = acc2[v];
  }
  __syncthreads();
  if (tid < N) {                                     // LN -> leaky -> logit
    const float* c = Csh + tid * H;
    float mu = 0.f;
    for (int h = 0; h < H; ++h) mu += c[h] + b_a[h];
    mu *= (1.f / H);
    float var = 0.f;
    for (int h = 0; h < H; ++h) { float d0 = c[h] + b_a[h] - mu; var += d0 * d0; }
    float rs = rsqrtf(var * (1.f / H) + 1e-5f);
    float lg = 0.f;
    for (int h = 0; h < H; ++h) {
      float y = (c[h] + b_a[h] - mu) * rs * ln_g[h] + ln_b[h];
      y = (y >= 0.f) ? y : 0.01f * y;
      lg += y * W_d[h];
    }
    lg += b_d[0];
    size_t off = ((size_t)b * N + i) * N + tid;
    float m = rel_mask[off];
    logits[tid] = (lg + sem_adj[off]) * m + (1.f - m) * (-1e30f);
  }
  __syncthreads();
  if (tid == 0) {
    float mx = -3.4e38f;
    for (int j = 0; j < N; ++j) mx = fmaxf(mx, logits[j]);
    sred[0] = mx;
  }
  __syncthreads();
  if (tid < N) logits[tid] = __expf(logits[tid] - sred[0]);
  __syncthreads();
  if (tid == 0) {
    float s = 0.f;
    for (int j = 0; j < N; ++j) s += logits[j];
    sred[1] = 1.f / s;
  }
  __syncthreads();
  if (tid < N) logits[tid] *= sred[1];               // attn row
  __syncthreads();
  const float* Fb = F32 + (size_t)b * N * D;
  for (int d = tid; d < D; d += 256) {
    float s = 0.f;
    for (int j = 0; j < N; ++j) s += logits[j] * Fb[(size_t)j * D + d];
    size_t off = ((size_t)b * N + i) * D + d;
    rf32[off] = s;
    rf_h[off] = f2bf(s);
  }
}

// ---- adj_out = rf*adj_diag + sum_j mr * rf_j * (1 - sig(qi+qj+mr*EK[r]+bk))
__global__ void __launch_bounds__(256)
k_adjout(const float* __restrict__ rf32, const float* __restrict__ qrf,
         const float* __restrict__ EK, const float* __restrict__ b_k,
         const float* __restrict__ adj, const float* __restrict__ rel_mask,
         const int* __restrict__ rel_adj, u16* __restrict__ adjout_h) {
  const int N = NNODE, D = DDIM;
  size_t t = (size_t)blockIdx.x * 256 + threadIdx.x;
  if (t >= (size_t)BATCH * N * D) return;
  int d = (int)(t % D);
  size_t bi = t / D;
  int b = (int)(bi / N), i = (int)(bi % N);
  float qi = qrf[bi * D + d], bk = b_k[d];
  size_t bb = (size_t)b * N * N;
  float acc = rf32[bi * D + d] * adj[bb + (size_t)i * N + i];
  for (int j = 0; j < N; ++j) {
    if (j == i) continue;
    float m = rel_mask[bb + (size_t)j * N + i];
    if (m == 0.f) continue;
    float mr = m * m;
    int r = rel_adj[bb + (size_t)j * N + i];
    float v = qi + qrf[((size_t)b * N + j) * D + d] + mr * EK[(size_t)r * D + d] + bk;
    float s = 1.f / (1.f + __expf(-v));
    acc += mr * rf32[((size_t)b * N + j) * D + d] * (1.f - s);
  }
  adjout_h[bi * D + d] = f2bf(acc);
}

// ---------------------------------------------------------------------------
static char* carve(char*& p, size_t bytes) {
  char* r = p;
  p += (bytes + 255) & ~(size_t)255;
  return r;
}

extern "C" void kernel_launch(void* const* d_in, const int* in_sizes, int n_in,
                              void* d_out, int out_size, void* d_ws,
                              size_t ws_size, hipStream_t stream) {
  const int B = BATCH, N = NNODE, D = DDIM;
  const float* adj      = (const float*)d_in[0];
  const int*   rel_adj  = (const int*)d_in[1];
  const float* feature  = (const float*)d_in[2];
  const float* rel_mask = (const float*)d_in[3];
  const int*   oneshot  = (const int*)d_in[4];
  const float* Edep     = (const float*)d_in[5];
  const float* Epos     = (const float*)d_in[6];
  const float* W_a = (const float*)d_in[7],  *b_a = (const float*)d_in[8];
  const float* ln_g = (const float*)d_in[9], *ln_b = (const float*)d_in[10];
  const float* W_d = (const float*)d_in[11], *b_d = (const float*)d_in[12];
  const float* W_f = (const float*)d_in[13], *b_f = (const float*)d_in[14];
  const float* W_q = (const float*)d_in[15], *b_q = (const float*)d_in[16];
  const float* W_k = (const float*)d_in[17], *b_k = (const float*)d_in[18];
  const float* W_lin = (const float*)d_in[19],  *b_lin = (const float*)d_in[20];
  const float* W_lin3 = (const float*)d_in[21], *b_lin3 = (const float*)d_in[22];

  char* p = (char*)d_ws;
  const int ND = B * N * D;                       // 294912
  float* F32     = (float*)carve(p, (size_t)ND * 4);
  u16*   F_h     = (u16*)carve(p, (size_t)ND * 2);
  u16*   sem_h   = (u16*)carve(p, (size_t)ND * 2);
  float* sem_adj = (float*)carve(p, (size_t)B * N * N * 4);
  float* EK      = (float*)carve(p, (size_t)50 * D * 4);
  u16*   Wf_h    = (u16*)carve(p, (size_t)D * D * 2);
  u16*   Wa_h    = (u16*)carve(p, (size_t)HDIM * D * 2);
  u16*   Wq_h    = (u16*)carve(p, (size_t)D * D * 2);
  u16*   Wl_h    = (u16*)carve(p, (size_t)D * D * 2);
  u16*   Wl3_h   = (u16*)carve(p, (size_t)256 * D * 2);
  u16*   tt_h    = (u16*)carve(p, (size_t)ND * 2);
  float* rf32    = (float*)carve(p, (size_t)ND * 4);
  u16*   rf_h    = (u16*)carve(p, (size_t)ND * 2);
  float* qrf32   = (float*)carve(p, (size_t)ND * 4);
  u16*   ao_h    = (u16*)carve(p, (size_t)ND * 2);
  (void)ws_size; (void)in_sizes; (void)n_in; (void)out_size;

  auto cvt = [&](const float* s, u16* d, int n) {
    k_cvt_bf16<<<(n + 255) / 256, 256, 0, stream>>>(s, d, n);
  };
  cvt(W_f, Wf_h, D * D);
  cvt(W_a, Wa_h, HDIM * D);
  cvt(W_q, Wq_h, D * D);
  cvt(W_lin, Wl_h, D * D);
  cvt(W_lin3, Wl3_h, 256 * D);
  cvt(feature, F_h, ND);
  hipMemcpyAsync(F32, feature, (size_t)ND * 4, hipMemcpyDeviceToDevice, stream);

  k_sem_rows<<<B * N, 256, 0, stream>>>(feature, sem_h);
  k_gram<<<36, 128, 0, stream>>>(sem_h, sem_adj);
  k_ek<<<(50 * D + 255) / 256, 256, 0, stream>>>(Edep, W_k, EK);

  // macro-tile grids: tiles = (M/32)*(N/32), 4 waves (tiles) per 128-thr block
  const int g768 = ((384 / 32) * (768 / 32) + 3) / 4;   // 72 blocks
  const int g256 = ((384 / 32) * (256 / 32) + 3) / 4;   // 24 blocks
  for (int it = 0; it < 2; ++it) {
    // tt = F @ W_f^T + b_f   (bf16 out)
    k_gemm<<<g768, 128, 0, stream>>>(F_h, Wf_h, b_f, nullptr, tt_h,
                                     B * N, D, D, 0);
    k_attn_rf<<<B * N, 256, 0, stream>>>(tt_h, Epos, oneshot, rel_adj, Edep,
                                         Wa_h, b_a, ln_g, ln_b, W_d, b_d,
                                         rel_mask, sem_adj, F32, rf32, rf_h);
    // qrf = rf @ W_q^T + b_q (f32 out)
    k_gemm<<<g768, 128, 0, stream>>>(rf_h, Wq_h, b_q, qrf32, nullptr,
                                     B * N, D, D, 0);
    k_adjout<<<(ND + 255) / 256, 256, 0, stream>>>(rf32, qrf32, EK, b_k, adj,
                                                   rel_mask, rel_adj, ao_h);
    if (it == 0) {
      // F = relu(adj_out @ W_lin^T + b_lin): f32 + bf16
      k_gemm<<<g768, 128, 0, stream>>>(ao_h, Wl_h, b_lin, F32, F_h,
                                       B * N, D, D, 1);
    } else {
      // out = adj_out @ W_lin3^T + b_lin3 -> (B,N,256) f32
      k_gemm<<<g256, 128, 0, stream>>>(ao_h, Wl3_h, b_lin3, (float*)d_out,
                                       nullptr, B * N, 256, D, 0);
    }
  }
}
